// ActorNetwork_6210522710220
// MI455X (gfx1250) — matrix operand order repaired and checked by reading, not compiled, not run on hardware
//
#include <hip/hip_runtime.h>
#include <hip/hip_bf16.h>

typedef __attribute__((ext_vector_type(16))) _Float16 v16h;
typedef __attribute__((ext_vector_type(8)))  _Float16 v8h;
typedef __attribute__((ext_vector_type(8)))  float    v8f;

#define HID 64
#define GATES 256          // 4*HID
#define ROWS_PER_BLOCK 128
#define THREADS 256

__device__ __forceinline__ float sigf(float x) {
    x = fminf(fmaxf(x, -30.f), 30.f);
    return 1.f / (1.f + exp2f(-1.44269504f * x));
}
__device__ __forceinline__ float tanh_fast(float x) {
    x = fminf(fmaxf(x, -15.f), 15.f);
    float e = exp2f(2.88539008f * x);   // e = exp(2x)
    return (e - 1.f) / (e + 1.f);
}

__global__ __launch_bounds__(THREADS) void lstm_actor_kernel(
    const float* __restrict__ x,
    const float* __restrict__ Wih0, const float* __restrict__ Whh0,
    const float* __restrict__ bih0, const float* __restrict__ bhh0,
    const float* __restrict__ Wih1, const float* __restrict__ Whh1,
    const float* __restrict__ bih1, const float* __restrict__ bhh1,
    const float* __restrict__ W1, const float* __restrict__ b1,
    const float* __restrict__ W2, const float* __restrict__ b2,
    const float* __restrict__ W3, const float* __restrict__ b3,
    float* __restrict__ out)
{
    // f16 weight copies for WMMA B-operands (row-major [256][64], K contiguous)
    __shared__ __attribute__((aligned(32))) _Float16 sWhh0[GATES * HID]; // 32KB
    __shared__ __attribute__((aligned(32))) _Float16 sWih1[GATES * HID]; // 32KB
    __shared__ __attribute__((aligned(32))) _Float16 sWhh1[GATES * HID]; // 32KB
    __shared__ __attribute__((aligned(32))) _Float16 sH[8 * 16 * HID];   // 16KB: per-wave h round-trip
    __shared__ float sB0[GATES], sB1[GATES];       // fused biases
    __shared__ float sWi0[GATES * 2];              // layer-0 input weights (K=2, scalar path)
    __shared__ float sX[ROWS_PER_BLOCK * 8];       // sequence inputs

    const int tid = threadIdx.x;
    const int blockBase = blockIdx.x * ROWS_PER_BLOCK;

    // ---- stage weights (f32 -> f16) and inputs into LDS ----
    for (int i = tid; i < GATES * HID; i += THREADS) {
        sWhh0[i] = (_Float16)Whh0[i];
        sWih1[i] = (_Float16)Wih1[i];
        sWhh1[i] = (_Float16)Whh1[i];
    }
    for (int i = tid; i < GATES; i += THREADS) {
        sB0[i] = bih0[i] + bhh0[i];
        sB1[i] = bih1[i] + bhh1[i];
    }
    for (int i = tid; i < GATES * 2; i += THREADS) sWi0[i] = Wih0[i];
    for (int i = tid; i < ROWS_PER_BLOCK * 8; i += THREADS) {
        int r = i >> 3, c = i & 7;
        sX[i] = x[(size_t)(blockBase + r) * 13 + c];
    }
    __syncthreads();

    const int lane = tid & 31;
    const int wid  = tid >> 5;
    const int nIdx = lane & 15;     // N within tile (B/C/D), row M for A frag
    const int half = lane >> 4;     // lane half selects K sub-range per ISA layouts
    const int mOff = half * 8;      // C/D rows held by this lane: M = mOff + vgpr
    _Float16* myH  = &sH[wid * 16 * HID];

    // 16x32 f16 A-fragment (M=batch rows, K=hidden) from row-major LDS, stride HID
    auto loadA = [&](const _Float16* base, int kc) -> v16h {
        const _Float16* p0 = base + nIdx * HID + kc * 32 + half * 16;
        return *(const v16h*)p0;         // K = kc*32 + half*16 .. +15
    };
    // 32x16 f16 B-fragment: B[k][n] = W[n][k]; per lane 16 contiguous K halves
    auto loadB = [&](const _Float16* W, int nt, int kc) -> v16h {
        const _Float16* p = W + (nt * 16 + nIdx) * HID + kc * 32 + half * 16;
        return *(const v16h*)p;
    };

    v16h h0A[2], h1A[2];
    v8f  c0[4], c1[4];
    {
        v16h zh;
        #pragma unroll
        for (int i = 0; i < 16; ++i) zh[i] = (_Float16)0.f;
        h0A[0] = zh; h0A[1] = zh; h1A[0] = zh; h1A[1] = zh;
        v8f zf;
        #pragma unroll
        for (int i = 0; i < 8; ++i) zf[i] = 0.f;
        #pragma unroll
        for (int n = 0; n < 4; ++n) { c0[n] = zf; c1[n] = zf; }
    }

    for (int t = 0; t < 4; ++t) {
        float xa[8], xb[8];
        #pragma unroll
        for (int e = 0; e < 8; ++e) {
            int rl = wid * 16 + mOff + e;
            xa[e] = sX[rl * 8 + 2 * t];
            xb[e] = sX[rl * 8 + 2 * t + 1];
        }
        // ---------------- LSTM layer 0 ----------------
        #pragma unroll
        for (int np = 0; np < 4; ++np) {
            v8f acc[4];
            #pragma unroll
            for (int gi = 0; gi < 4; ++gi) {
                int col = gi * 64 + np * 16 + nIdx;
                float bv = sB0[col];
                float wa = sWi0[2 * col], wb = sWi0[2 * col + 1];
                v8f a;
                #pragma unroll
                for (int e = 0; e < 8; ++e) a[e] = fmaf(wa, xa[e], fmaf(wb, xb[e], bv));
                #pragma unroll
                for (int kc = 0; kc < 2; ++kc) {
                    v16h Bf = loadB(sWhh0, gi * 4 + np, kc);
                    a = __builtin_amdgcn_wmma_f32_16x16x32_f16(false, h0A[kc], false, Bf, (short)0, a, false, false);
                }
                acc[gi] = a;
            }
            v8f cn;
            #pragma unroll
            for (int e = 0; e < 8; ++e) {
                float iv = sigf(acc[0][e]);
                float fv = sigf(acc[1][e]);
                float gv = tanh_fast(acc[2][e]);
                float ov = sigf(acc[3][e]);
                float cc = fmaf(fv, c0[np][e], iv * gv);
                cn[e] = cc;
                float hh = ov * tanh_fast(cc);
                myH[(mOff + e) * HID + np * 16 + nIdx] = (_Float16)hh;
            }
            c0[np] = cn;
        }
        asm volatile("s_wait_dscnt 0" ::: "memory");   // per-wave LDS RAW ordering
        h0A[0] = loadA(myH, 0);
        h0A[1] = loadA(myH, 1);
        // ---------------- LSTM layer 1 ----------------
        #pragma unroll
        for (int np = 0; np < 4; ++np) {
            v8f acc[4];
            #pragma unroll
            for (int gi = 0; gi < 4; ++gi) {
                int col = gi * 64 + np * 16 + nIdx;
                float bv = sB1[col];
                v8f a;
                #pragma unroll
                for (int e = 0; e < 8; ++e) a[e] = bv;
                #pragma unroll
                for (int kc = 0; kc < 2; ++kc) {
                    v16h Bi = loadB(sWih1, gi * 4 + np, kc);
                    a = __builtin_amdgcn_wmma_f32_16x16x32_f16(false, h0A[kc], false, Bi, (short)0, a, false, false);
                    v16h Bh = loadB(sWhh1, gi * 4 + np, kc);
                    a = __builtin_amdgcn_wmma_f32_16x16x32_f16(false, h1A[kc], false, Bh, (short)0, a, false, false);
                }
                acc[gi] = a;
            }
            v8f cn;
            #pragma unroll
            for (int e = 0; e < 8; ++e) {
                float iv = sigf(acc[0][e]);
                float fv = sigf(acc[1][e]);
                float gv = tanh_fast(acc[2][e]);
                float ov = sigf(acc[3][e]);
                float cc = fmaf(fv, c1[np][e], iv * gv);
                cn[e] = cc;
                float hh = ov * tanh_fast(cc);
                myH[(mOff + e) * HID + np * 16 + nIdx] = (_Float16)hh;
            }
            c1[np] = cn;
        }
        asm volatile("s_wait_dscnt 0" ::: "memory");
        h1A[0] = loadA(myH, 0);
        h1A[1] = loadA(myH, 1);
    }

    __syncthreads();   // sH now holds final h1 for all 128 rows

    // ---------------- MLP head: one thread per row ----------------
    if (tid < ROWS_PER_BLOCK) {
        const int row = blockBase + tid;
        float d[69];
        #pragma unroll
        for (int k = 0; k < 64; ++k) d[k] = (float)sH[tid * HID + k];
        #pragma unroll
        for (int j = 0; j < 5; ++j) d[64 + j] = x[(size_t)row * 13 + 8 + j];

        float l1v[30];
        #pragma unroll
        for (int n = 0; n < 30; ++n) {
            float s = b1[n];
            #pragma unroll
            for (int k = 0; k < 69; ++k) s = fmaf(W1[n * 69 + k], d[k], s);
            l1v[n] = fmaxf(s, 0.f);
        }
        float l2v[10];
        #pragma unroll
        for (int n = 0; n < 10; ++n) {
            float s = b2[n];
            #pragma unroll
            for (int k = 0; k < 30; ++k) s = fmaf(W2[n * 30 + k], l1v[k], s);
            l2v[n] = fmaxf(s, 0.f);
        }
        float l3v[4];
        #pragma unroll
        for (int n = 0; n < 4; ++n) {
            float s = b3[n];
            #pragma unroll
            for (int k = 0; k < 10; ++k) s = fmaf(W3[n * 10 + k], l2v[k], s);
            l3v[n] = s;
        }
        float m = fmaxf(fmaxf(l3v[0], l3v[1]), fmaxf(l3v[2], l3v[3]));
        float se = 0.f;
        #pragma unroll
        for (int n = 0; n < 4; ++n) se += exp2f((l3v[n] - m) * 1.44269504f);
        float lse = log2f(se) * 0.69314718f;
        #pragma unroll
        for (int n = 0; n < 4; ++n) out[(size_t)row * 4 + n] = l3v[n] - m - lse;
    }
}

extern "C" void kernel_launch(void* const* d_in, const int* in_sizes, int n_in,
                              void* d_out, int out_size, void* d_ws, size_t ws_size,
                              hipStream_t stream) {
    const float* x    = (const float*)d_in[0];
    const float* Wih0 = (const float*)d_in[1];
    const float* Whh0 = (const float*)d_in[2];
    const float* bih0 = (const float*)d_in[3];
    const float* bhh0 = (const float*)d_in[4];
    const float* Wih1 = (const float*)d_in[5];
    const float* Whh1 = (const float*)d_in[6];
    const float* bih1 = (const float*)d_in[7];
    const float* bhh1 = (const float*)d_in[8];
    const float* W1   = (const float*)d_in[9];
    const float* b1   = (const float*)d_in[10];
    const float* W2   = (const float*)d_in[11];
    const float* b2   = (const float*)d_in[12];
    const float* W3   = (const float*)d_in[13];
    const float* b3   = (const float*)d_in[14];

    const int rows   = in_sizes[0] / 13;           // 131072
    const int blocks = rows / ROWS_PER_BLOCK;      // 1024

    lstm_actor_kernel<<<dim3(blocks), dim3(THREADS), 0, stream>>>(
        x, Wih0, Whh0, bih0, bhh0, Wih1, Whh1, bih1, bhh1,
        W1, b1, W2, b2, W3, b3, (float*)d_out);
}
